// BiMoEMambaTrajectory_53515292508676
// MI455X (gfx1250) — matrix operand order repaired and checked by reading, not compiled, not run on hardware
//
#include <hip/hip_runtime.h>
#include <math.h>

// ---------------- model dims ----------------
constexpr int BB = 128, TT = 64, FIN = 9, DD = 256, LL = 6;
constexpr int DI = 512, DS_ = 16, DTR = 16, DFF = 1024, PP = 20, KK = 6, HH = 4;
constexpr size_t M_ENC = (size_t)BB * TT;          // 8192
constexpr size_t M_GATE = (size_t)BB * PP;         // 2560
constexpr size_t M_DEC = (size_t)BB * KK * PP;     // 15360

typedef __attribute__((ext_vector_type(16))) __bf16 v16bf;
typedef __attribute__((ext_vector_type(4)))  __bf16 v4bf;
typedef __attribute__((ext_vector_type(8)))  float  v8f;

enum { ACT_NONE = 0, ACT_SILU = 1, ACT_SOFTPLUS = 2, ACT_GELU = 3 };

__device__ __forceinline__ float dev_silu(float v) { return v / (1.f + expf(-v)); }
__device__ __forceinline__ float dev_softplus(float v) { return (v > 20.f) ? v : log1pf(expf(v)); }

// ======================================================================
// Generic WMMA GEMM:  C[M,N] = act(A[M,K] @ W[N,K]^T + bias)
// 256 threads = 8 waves; block tile 32(M) x 128(N); wave tile 16x32
// (2 accumulators, 2 v_wmma_f32_16x16x32_bf16 per K-chunk of 32).
// Fast path: float4 global loads -> packed bf16 -> 8B LDS stores,
// plus global_prefetch of the next K-chunk. Guarded path for edges.
// ======================================================================
__global__ void gemm_wmma(const float* __restrict__ A, int lda,
                          const float* __restrict__ W, int ldw,
                          const float* __restrict__ bias,
                          float* __restrict__ C, int ldc,
                          int M, int N, int K, int act)
{
    __shared__ __bf16 As[32][32];
    __shared__ __bf16 Bs[128][32];

    const int tid  = threadIdx.x;
    const int wave = tid >> 5;
    const int lane = tid & 31;
    const int half = lane >> 4;
    const int l16  = lane & 15;
    const int wm   = wave >> 2;          // 0..1  (M sub-tile)
    const int wn   = wave & 3;           // 0..3  (N sub-tile)
    const int m0 = blockIdx.y * 32;
    const int n0 = blockIdx.x * 128;

    const bool aFast = ((lda & 3) == 0) && (m0 + 32 <= M);
    const bool bFast = ((ldw & 3) == 0) && (n0 + 128 <= N);

    v8f acc0 = {}, acc1 = {};

    const int ar = tid >> 3;             // 0..31 row within A tile
    const int ak = (tid & 7) * 4;        // K quad

    for (int k0 = 0; k0 < K; k0 += 32) {
        const bool kFull = (k0 + 32 <= K);

        // ---- stage A tile (32x32) ----
        if (aFast && kFull) {
            const float* ap = A + (size_t)(m0 + ar) * lda + k0 + ak;
            float4 v = *(const float4*)ap;
            v4bf o = { (__bf16)v.x, (__bf16)v.y, (__bf16)v.z, (__bf16)v.w };
            *(v4bf*)&As[ar][ak] = o;
            if (k0 + 32 < K) __builtin_prefetch(ap + 32, 0, 1);
        } else {
            for (int i = tid; i < 32 * 32; i += 256) {
                int r = i >> 5, kk = i & 31;
                int gr = m0 + r, gk = k0 + kk;
                float v = (gr < M && gk < K) ? A[(size_t)gr * lda + gk] : 0.f;
                As[r][kk] = (__bf16)v;
            }
        }
        // ---- stage W tile (128x32) ----
        if (bFast && kFull) {
#pragma unroll
            for (int j = 0; j < 4; ++j) {
                int c = tid + 256 * j;
                int r = c >> 3, kk = (c & 7) * 4;
                const float* wp = W + (size_t)(n0 + r) * ldw + k0 + kk;
                float4 v = *(const float4*)wp;
                v4bf o = { (__bf16)v.x, (__bf16)v.y, (__bf16)v.z, (__bf16)v.w };
                *(v4bf*)&Bs[r][kk] = o;
                if (k0 + 32 < K) __builtin_prefetch(wp + 32, 0, 1);
            }
        } else {
            for (int i = tid; i < 128 * 32; i += 256) {
                int r = i >> 5, kk = i & 31;
                int gn = n0 + r, gk = k0 + kk;
                float v = (gn < N && gk < K) ? W[(size_t)gn * ldw + gk] : 0.f;
                Bs[r][kk] = (__bf16)v;
            }
        }
        __syncthreads();

        // A fragment: 16-bit A 16x32 lane layout (ISA 7.12.2)
        v16bf afrag, b0, b1;
#pragma unroll
        for (int e = 0; e < 16; ++e) {
            int v = e >> 1, p = e & 1;
            int kk = ((v < 4) ? 0 : 16) + (v & 3) * 2 + p + half * 8;
            afrag[e] = As[wm * 16 + l16][kk];
        }
        // B fragments: lanes 0-15 K=0..7,16..23, lanes 16-31 K=8..15,24..31
#pragma unroll
        for (int e = 0; e < 16; ++e) {
            b0[e] = Bs[wn * 32 + l16][e + 8 * half + ((e >> 3) << 3)];
            b1[e] = Bs[wn * 32 + 16 + l16][e + 8 * half + ((e >> 3) << 3)];
        }
        acc0 = __builtin_amdgcn_wmma_f32_16x16x32_bf16(
            false, afrag, false, b0, (short)0, acc0, false, false);
        acc1 = __builtin_amdgcn_wmma_f32_16x16x32_bf16(
            false, afrag, false, b1, (short)0, acc1, false, false);
        __syncthreads();
    }

    // epilogue: C layout: VGPR r -> M = r + 8*half, N = lane&15
#pragma unroll
    for (int s = 0; s < 2; ++s) {
        int cn = n0 + wn * 32 + s * 16 + l16;
        if (cn < N) {
            float bv = bias ? bias[cn] : 0.f;
#pragma unroll
            for (int r = 0; r < 8; ++r) {
                int cm = m0 + wm * 16 + r + 8 * half;
                if (cm < M) {
                    float v = (s == 0 ? acc0[r] : acc1[r]) + bv;
                    if (act == ACT_SILU)          v = dev_silu(v);
                    else if (act == ACT_SOFTPLUS) v = dev_softplus(v);
                    else if (act == ACT_GELU)     v = 0.5f * v * (1.f + erff(v * 0.70710678f));
                    C[(size_t)cm * ldc + cn] = v;
                }
            }
        }
    }
}

// ============================ elementwise / small kernels ============================

// h = x @ in_w^T + in_b + PE
__global__ void k_in_proj(const float* __restrict__ x, const float* __restrict__ in_w,
                          const float* __restrict__ in_b, float* __restrict__ h)
{
    size_t idx = (size_t)blockIdx.x * blockDim.x + threadIdx.x;  // M_ENC*256
    int d = (int)(idx & 255);
    size_t row = idx >> 8;
    int t = (int)(row & 63);
    const float* xr = x + row * FIN;
    const float* wr = in_w + (size_t)d * FIN;
    float s = in_b[d];
#pragma unroll
    for (int f = 0; f < FIN; ++f) s += xr[f] * wr[f];
    int i = d >> 1;
    float div = expf((float)(2 * i) * (-9.210340371976184f / 256.f));
    float ang = (float)t * div;
    s += (d & 1) ? cosf(ang) : sinf(ang);
    h[idx] = s;
}

// gate_feat[b][c] = mean over last 4 timesteps of x[b,t,7+c]
__global__ void k_gatefeat(const float* __restrict__ x, float* __restrict__ gf)
{
    int i = blockIdx.x * blockDim.x + threadIdx.x;
    if (i >= 2 * BB) return;
    int c = i & 1, b = i >> 1;
    float s = 0.f;
    for (int j = 0; j < 4; ++j) s += x[((size_t)b * TT + 60 + j) * FIN + 7 + c];
    gf[b * 2 + c] = s * 0.25f;
}

// AdaLN conditioning: per batch compute mult (scale+tanh(g)*.5) and add (tanh(b)*.5)
__global__ void k_ada_cond(const float* __restrict__ x, const float* __restrict__ scale,
                           const float* __restrict__ cw1, const float* __restrict__ cb1,
                           const float* __restrict__ cw2, const float* __restrict__ cb2,
                           float* __restrict__ mod)
{
    __shared__ float hc1[512];
    int b = blockIdx.x;
    float p0 = x[((size_t)b * TT + 63) * FIN + 5];
    float p1 = x[((size_t)b * TT + 63) * FIN + 6];
    for (int j = threadIdx.x; j < 512; j += blockDim.x) {
        float v = cw1[j * 2] * p0 + cw1[j * 2 + 1] * p1 + cb1[j];
        hc1[j] = dev_silu(v);
    }
    __syncthreads();
    for (int i = threadIdx.x; i < 512; i += blockDim.x) {
        float s = cb2[i];
        const float* w = cw2 + (size_t)i * 512;
        for (int j = 0; j < 512; ++j) s += hc1[j] * w[j];
        float tv = tanhf(s) * 0.5f;
        mod[(size_t)b * 512 + i] = (i < 256) ? (scale[i] + tv) : tv;
    }
}

// u = rmsnorm(h) * mult + add  (block per row, 256 threads)
__global__ void k_rmsmod(const float* __restrict__ h, const float* __restrict__ mod,
                         float* __restrict__ u)
{
    __shared__ float s[256];
    size_t row = blockIdx.x;
    int d = threadIdx.x, b = (int)(row >> 6);
    float v = h[row * 256 + d];
    s[d] = v * v;
    __syncthreads();
    for (int st = 128; st > 0; st >>= 1) { if (d < st) s[d] += s[d + st]; __syncthreads(); }
    float rinv = rsqrtf(s[0] / 256.f + 1e-6f);
    u[row * 256 + d] = v * rinv * mod[(size_t)b * 512 + d] + mod[(size_t)b * 512 + 256 + d];
}

// depthwise causal (dir=0) / anti-causal (dir=1) conv + SiLU
__global__ void k_conv(const float* __restrict__ xz, const float* __restrict__ cw,
                       const float* __restrict__ cb, float* __restrict__ xc, int dir)
{
    size_t idx = (size_t)blockIdx.x * blockDim.x + threadIdx.x;
    if (idx >= M_ENC * DI) return;
    int d = (int)(idx & 511);
    size_t row = idx >> 9;
    int t = (int)(row & 63), b = (int)(row >> 6);
    float s = cb[d];
#pragma unroll
    for (int j = 0; j < 4; ++j) {
        int tt = dir ? (t + 3 - j) : (t + j - 3);
        if (tt >= 0 && tt < TT)
            s += cw[d * 4 + j] * xz[((size_t)(b * TT + tt)) * 1024 + d];
    }
    xc[idx] = dev_silu(s);
}

// selective-scan: block per batch, 512 threads (one per channel), state in VGPRs
__global__ void k_scan(const float* __restrict__ xz, const float* __restrict__ xc,
                       const float* __restrict__ dbc, const float* __restrict__ dtb,
                       const float* __restrict__ alog, const float* __restrict__ dsk,
                       float* __restrict__ ym, int dir)
{
    __shared__ float bs[16], cs[16];
    int b = blockIdx.x, d = threadIdx.x;
    float a[16], st[16];
#pragma unroll
    for (int s = 0; s < 16; ++s) { a[s] = -expf(alog[d * 16 + s]); st[s] = 0.f; }
    float dskip = dsk[d];
    for (int i = 0; i < TT; ++i) {
        int t = dir ? (TT - 1 - i) : i;
        size_t row = (size_t)b * TT + t;
        if (threadIdx.x < 16) bs[threadIdx.x] = dbc[row * 48 + 16 + threadIdx.x];
        else if (threadIdx.x < 32) cs[threadIdx.x - 16] = dbc[row * 48 + 32 + (threadIdx.x - 16)];
        __syncthreads();
        float dt = dtb[row * 512 + d];
        float xv = xc[row * 512 + d];
        float y = 0.f;
#pragma unroll
        for (int s = 0; s < 16; ++s) {
            st[s] = expf(dt * a[s]) * st[s] + dt * xv * bs[s];
            y += st[s] * cs[s];
        }
        y += dskip * xv;
        float z = xz[row * 1024 + 512 + d];
        ym[row * 512 + d] = y * dev_silu(z);
        __syncthreads();
    }
}

// h[i] += ls[i%256] * y[i]
__global__ void k_axpy(float* __restrict__ h, const float* __restrict__ y,
                       const float* __restrict__ ls, size_t n)
{
    size_t i = (size_t)blockIdx.x * blockDim.x + threadIdx.x;
    if (i < n) h[i] += ls[i & 255] * y[i];
}

// gated-FFN: out = silu(gate)*val  from [rows,2048] -> [rows,1024]
__global__ void k_glu(const float* __restrict__ ffh, float* __restrict__ out, size_t rows)
{
    size_t idx = (size_t)blockIdx.x * blockDim.x + threadIdx.x;
    if (idx >= rows * 1024) return;
    int j = (int)(idx & 1023);
    size_t row = idx >> 10;
    float g = ffh[row * 2048 + j];
    float v = ffh[row * 2048 + 1024 + j];
    out[idx] = dev_silu(g) * v;
}

// 4-head attention: q rows shared across batch; kv[b*T + t] = [K(256) | V(256)]
// block = (b*G + g)*P + p, 64 threads. Query row staged to LDS with the CDNA5
// async-to-LDS path (ASYNCcnt), f32 so no conversion needed.
__global__ void k_attn(const float* __restrict__ q, const float* __restrict__ kv,
                       float* __restrict__ out, int G, int P, float scale)
{
    __shared__ __align__(16) float qs[256];
    __shared__ float sc[4 * 64];
    int blk = blockIdx.x;
    int p = blk % P;
    int g = (blk / P) % G;
    int b = blk / (P * G);
    const float* qrow = q + (size_t)(g * P + p) * 256;
    {
        unsigned ldsoff = (unsigned)(size_t)(&qs[0]) + (unsigned)threadIdx.x * 16u;
        unsigned long long ga = (unsigned long long)(size_t)(qrow + threadIdx.x * 4);
        asm volatile("global_load_async_to_lds_b128 %0, %1, off"
                     :: "v"(ldsoff), "v"(ga) : "memory");
        asm volatile("s_wait_asynccnt 0x0" ::: "memory");
    }
    __syncthreads();
    int t = threadIdx.x;
    const float* kr = kv + ((size_t)b * TT + t) * 512;
#pragma unroll
    for (int hh = 0; hh < 4; ++hh) {
        float s = 0.f;
        for (int dd = 0; dd < 64; ++dd) s += qs[hh * 64 + dd] * kr[hh * 64 + dd];
        sc[hh * 64 + t] = s * scale;
    }
    __syncthreads();
    int dloc = threadIdx.x;
    float* orow = out + (size_t)blk * 256;
#pragma unroll
    for (int hh = 0; hh < 4; ++hh) {
        float m = -1e30f;
        for (int tt = 0; tt < TT; ++tt) m = fmaxf(m, sc[hh * 64 + tt]);
        float l = 0.f;
        for (int tt = 0; tt < TT; ++tt) l += expf(sc[hh * 64 + tt] - m);
        float acc = 0.f;
        for (int tt = 0; tt < TT; ++tt) {
            float w = expf(sc[hh * 64 + tt] - m);
            acc += w * kv[((size_t)b * TT + tt) * 512 + 256 + hh * 64 + dloc];
        }
        orow[hh * 64 + dloc] = acc / l;
    }
}

// out[row] = LN(base[row % baseMod] + ls*add[row]) * g + b   (block/row, 256 thr)
__global__ void k_addln(const float* __restrict__ base, int baseMod,
                        const float* __restrict__ ls, const float* __restrict__ add,
                        const float* __restrict__ gam, const float* __restrict__ bet,
                        float* __restrict__ out)
{
    __shared__ float s1[256];
    __shared__ float s2[256];
    size_t row = blockIdx.x;
    int d = threadIdx.x;
    float lsv = ls ? ls[d] : 1.f;
    float v = base[(size_t)(row % baseMod) * 256 + d] + lsv * add[row * 256 + d];
    s1[d] = v;
    __syncthreads();
    for (int st = 128; st > 0; st >>= 1) { if (d < st) s1[d] += s1[d + st]; __syncthreads(); }
    float mean = s1[0] / 256.f;
    float dv = v - mean;
    s2[d] = dv * dv;
    __syncthreads();
    for (int st = 128; st > 0; st >>= 1) { if (d < st) s2[d] += s2[d + st]; __syncthreads(); }
    float var = s2[0] / 256.f;
    out[row * 256 + d] = dv * rsqrtf(var + 1e-5f) * gam[d] + bet[d];
}

// concat [step_ctx | gate_feat] -> [B*P, 258]
__global__ void k_gmcat(const float* __restrict__ sctx, const float* __restrict__ gf,
                        float* __restrict__ out)
{
    size_t idx = (size_t)blockIdx.x * blockDim.x + threadIdx.x;
    if (idx >= M_GATE * 258) return;
    int col = (int)(idx % 258);
    size_t row = idx / 258;
    int b = (int)(row / PP);
    out[idx] = (col < 256) ? sctx[row * 256 + col] : gf[b * 2 + (col - 256)];
}

// logits_step[b,p,k] = hid[b,p] . gm_w2[k] + gm_b2[k]
__global__ void k_logits_step(const float* __restrict__ hid, const float* __restrict__ w,
                              const float* __restrict__ bias, float* __restrict__ out)
{
    size_t idx = (size_t)blockIdx.x * blockDim.x + threadIdx.x;
    if (idx >= M_GATE * KK) return;
    int k = (int)(idx % KK);
    size_t row = idx / KK;
    const float* hr = hid + row * 256;
    const float* wr = w + (size_t)k * 256;
    float s = bias[k];
    for (int j = 0; j < 256; ++j) s += hr[j] * wr[j];
    out[row * KK + k] = s;
}

__global__ void k_logits_mean(const float* __restrict__ lstep, float* __restrict__ out)
{
    int idx = blockIdx.x * blockDim.x + threadIdx.x;
    if (idx >= BB * KK) return;
    int k = idx % KK, b = idx / KK;
    float s = 0.f;
    for (int p = 0; p < PP; ++p) s += lstep[((size_t)b * PP + p) * KK + k];
    out[idx] = s / (float)PP;
}

// output heads: mu/sigma/rho
__global__ void k_heads(const float* __restrict__ h2, const float* __restrict__ hw,
                        const float* __restrict__ hb, const float* __restrict__ mean,
                        const float* __restrict__ stdv, float* __restrict__ out,
                        size_t o_mu, size_t o_sg, size_t o_rho)
{
    size_t idx = (size_t)blockIdx.x * blockDim.x + threadIdx.x;  // (b*K+k)*P+p
    if (idx >= M_DEC) return;
    int k = (int)((idx / PP) % KK);
    const float* hr = h2 + idx * 256;
    float raw[5];
#pragma unroll
    for (int o = 0; o < 5; ++o) {
        const float* wr = hw + (size_t)(k * 5 + o) * 256;
        float s = hb[k * 5 + o];
        for (int j = 0; j < 256; ++j) s += hr[j] * wr[j];
        raw[o] = s;
    }
    out[o_mu + idx * 2 + 0] = raw[0] * stdv[0] + mean[0];
    out[o_mu + idx * 2 + 1] = raw[1] * stdv[1] + mean[1];
    out[o_sg + idx * 2 + 0] = (dev_softplus(raw[2]) + 0.001f) * stdv[0];
    out[o_sg + idx * 2 + 1] = (dev_softplus(raw[3]) + 0.001f) * stdv[1];
    float r = tanhf(raw[4]);
    out[o_rho + idx] = fminf(0.999f, fmaxf(-0.999f, r));
}

// ============================ host orchestration ============================

static void launch_gemm(const float* A, int lda, const float* W, int ldw,
                        const float* bias, float* C, int ldc,
                        int M, int N, int K, int act, hipStream_t s)
{
    dim3 g((N + 127) / 128, (M + 31) / 32);
    gemm_wmma<<<g, 256, 0, s>>>(A, lda, W, ldw, bias, C, ldc, M, N, K, act);
}

// workspace layout (floats)
constexpr size_t OFF_H   = 0;
constexpr size_t OFF_U   = OFF_H + M_ENC * 256;
constexpr size_t OFF_MOD = OFF_U + M_ENC * 256;
constexpr size_t OFF_GF  = OFF_MOD + (size_t)BB * 512;
constexpr size_t OFF_Y   = OFF_GF + 256;
constexpr size_t POOL    = OFF_Y + M_ENC * 256;
// encoder-phase pool
constexpr size_t P_XZ  = POOL;
constexpr size_t P_XC  = P_XZ + M_ENC * 1024;
constexpr size_t P_DBC = P_XC + M_ENC * 512;
constexpr size_t P_DT  = P_DBC + M_ENC * 48;
constexpr size_t P_YM  = P_DT + M_ENC * 512;
constexpr size_t P_CAT = P_YM + M_ENC * 512;
constexpr size_t P_FFH = P_CAT + M_ENC * 512;
constexpr size_t P_FFT = P_FFH + M_ENC * 2048;
// decoder-phase pool (reuses encoder pool region)
constexpr size_t P_KVG  = POOL;
constexpr size_t P_QPG  = P_KVG + M_ENC * 512;
constexpr size_t P_AO   = P_QPG + 32 * 256;
constexpr size_t P_AO2  = P_AO + M_GATE * 256;
constexpr size_t P_SCTX = P_AO2 + M_GATE * 256;
constexpr size_t P_GCAT = P_SCTX + M_GATE * 256;
constexpr size_t P_HID  = P_GCAT + M_GATE * 258;
constexpr size_t P_KVC  = P_HID + M_GATE * 256;
constexpr size_t P_QPC  = P_KVC + M_ENC * 512;
constexpr size_t P_ATTC = P_QPC + 128 * 256;
constexpr size_t P_OC   = P_ATTC + M_DEC * 256;
constexpr size_t P_H2   = P_OC + M_DEC * 256;
constexpr size_t P_DFFH = P_H2 + M_DEC * 256;
constexpr size_t P_DFF  = P_DFFH + M_DEC * 1024;

extern "C" void kernel_launch(void* const* d_in, const int* in_sizes, int n_in,
                              void* d_out, int out_size, void* d_ws, size_t ws_size,
                              hipStream_t stream)
{
    (void)in_sizes; (void)n_in; (void)out_size; (void)ws_size;
    auto in = [&](int i) { return (const float*)d_in[i]; };
    float* ws = (float*)d_ws;
    float* out = (float*)d_out;

    const float* x = in(0);
    float* h   = ws + OFF_H;
    float* u   = ws + OFF_U;
    float* mod = ws + OFF_MOD;
    float* gf  = ws + OFF_GF;
    float* y   = ws + OFF_Y;

    // ---- input projection + positional encoding ----
    k_in_proj<<<(unsigned)(M_ENC * 256 / 256), 256, 0, stream>>>(x, in(1), in(2), h);
    k_gatefeat<<<1, 256, 0, stream>>>(x, gf);

    // ---- encoder layers ----
    for (int l = 0; l < LL; ++l) {
        // AdaRMSNorm 1
        k_ada_cond<<<BB, 256, 0, stream>>>(x, in(3) + l * 256, in(4) + l * 1024,
                                           in(5) + l * 512, in(6) + (size_t)l * 512 * 512,
                                           in(7) + l * 512, mod);
        k_rmsmod<<<(unsigned)M_ENC, 256, 0, stream>>>(h, mod, u);

        // bidirectional Mamba
        for (int dir = 0; dir < 2; ++dir) {
            size_t w = (size_t)(l * 2 + dir);
            launch_gemm(u, 256, in(13) + w * 1024 * 256, 256, nullptr,
                        ws + P_XZ, 1024, (int)M_ENC, 1024, 256, ACT_NONE, stream);
            k_conv<<<(unsigned)(M_ENC * 512 / 256), 256, 0, stream>>>(
                ws + P_XZ, in(14) + w * 512 * 4, in(15) + w * 512, ws + P_XC, dir);
            launch_gemm(ws + P_XC, 512, in(16) + w * 48 * 512, 512, nullptr,
                        ws + P_DBC, 48, (int)M_ENC, 48, 512, ACT_NONE, stream);
            launch_gemm(ws + P_DBC, 48, in(17) + w * 512 * 16, 16, in(18) + w * 512,
                        ws + P_DT, 512, (int)M_ENC, 512, 16, ACT_SOFTPLUS, stream);
            k_scan<<<BB, 512, 0, stream>>>(ws + P_XZ, ws + P_XC, ws + P_DBC, ws + P_DT,
                                           in(19) + w * 512 * 16, in(20) + w * 512,
                                           ws + P_YM, dir);
            launch_gemm(ws + P_YM, 512, in(21) + w * 256 * 512, 512, nullptr,
                        ws + P_CAT + dir * 256, 512, (int)M_ENC, 256, 512, ACT_NONE, stream);
        }
        launch_gemm(ws + P_CAT, 512, in(22) + (size_t)l * 256 * 512, 512, in(23) + l * 256,
                    y, 256, (int)M_ENC, 256, 512, ACT_NONE, stream);
        k_axpy<<<(unsigned)(M_ENC * 256 / 256), 256, 0, stream>>>(h, y, in(28) + l * 256,
                                                                  M_ENC * 256);

        // AdaRMSNorm 2 + gated FFN
        k_ada_cond<<<BB, 256, 0, stream>>>(x, in(8) + l * 256, in(9) + l * 1024,
                                           in(10) + l * 512, in(11) + (size_t)l * 512 * 512,
                                           in(12) + l * 512, mod);
        k_rmsmod<<<(unsigned)M_ENC, 256, 0, stream>>>(h, mod, u);
        launch_gemm(u, 256, in(24) + (size_t)l * 2048 * 256, 256, in(25) + l * 2048,
                    ws + P_FFH, 2048, (int)M_ENC, 2048, 256, ACT_NONE, stream);
        k_glu<<<(unsigned)(M_ENC * 1024 / 256), 256, 0, stream>>>(ws + P_FFH, ws + P_FFT, M_ENC);
        launch_gemm(ws + P_FFT, 1024, in(26) + (size_t)l * 256 * 1024, 1024, in(27) + l * 256,
                    y, 256, (int)M_ENC, 256, 1024, ACT_NONE, stream);
        k_axpy<<<(unsigned)(M_ENC * 256 / 256), 256, 0, stream>>>(h, y, in(29) + l * 256,
                                                                  M_ENC * 256);
    }

    // ---- gate attention head ----
    launch_gemm(in(30), 256, in(31), 256, in(32), ws + P_QPG, 256, PP, 256, 256, ACT_NONE, stream);
    launch_gemm(h, 256, in(31) + 256 * 256, 256, in(32) + 256,
                ws + P_KVG, 512, (int)M_ENC, 512, 256, ACT_NONE, stream);
    k_attn<<<(unsigned)(BB * PP), 64, 0, stream>>>(ws + P_QPG, ws + P_KVG, ws + P_AO,
                                                   1, PP, 0.125f);
    launch_gemm(ws + P_AO, 256, in(33), 256, in(34), ws + P_AO2, 256,
                (int)M_GATE, 256, 256, ACT_NONE, stream);
    k_addln<<<(unsigned)M_GATE, 256, 0, stream>>>(in(30), PP, nullptr, ws + P_AO2,
                                                  in(35), in(36), ws + P_SCTX);
    k_gmcat<<<(unsigned)((M_GATE * 258 + 255) / 256), 256, 0, stream>>>(ws + P_SCTX, gf,
                                                                        ws + P_GCAT);
    launch_gemm(ws + P_GCAT, 258, in(37), 258, in(38), ws + P_HID, 256,
                (int)M_GATE, 256, 258, ACT_SILU, stream);
    k_logits_step<<<(unsigned)((M_GATE * KK + 255) / 256), 256, 0, stream>>>(
        ws + P_HID, in(39), in(40), out + BB * KK);
    k_logits_mean<<<(BB * KK + 255) / 256, 256, 0, stream>>>(out + BB * KK, out);

    // ---- cross-attention decoder ----
    launch_gemm(in(41), 256, in(42), 256, in(43), ws + P_QPC, 256,
                KK * PP, 256, 256, ACT_NONE, stream);
    launch_gemm(h, 256, in(42) + 256 * 256, 256, in(43) + 256,
                ws + P_KVC, 512, (int)M_ENC, 512, 256, ACT_NONE, stream);
    k_attn<<<(unsigned)(BB * KK * PP), 64, 0, stream>>>(ws + P_QPC, ws + P_KVC, ws + P_ATTC,
                                                        KK, PP, 0.125f);
    launch_gemm(ws + P_ATTC, 256, in(44), 256, in(45), ws + P_OC, 256,
                (int)M_DEC, 256, 256, ACT_NONE, stream);
    k_addln<<<(unsigned)M_DEC, 256, 0, stream>>>(in(41), KK * PP, in(54), ws + P_OC,
                                                 in(46), in(47), ws + P_H2);
    launch_gemm(ws + P_H2, 256, in(50), 256, in(51), ws + P_DFFH, 1024,
                (int)M_DEC, 1024, 256, ACT_GELU, stream);
    launch_gemm(ws + P_DFFH, 1024, in(52), 1024, in(53), ws + P_DFF, 256,
                (int)M_DEC, 256, 1024, ACT_NONE, stream);
    k_addln<<<(unsigned)M_DEC, 256, 0, stream>>>(ws + P_H2, (int)M_DEC, in(55), ws + P_DFF,
                                                 in(48), in(49), ws + P_H2);

    // ---- output heads ----
    const size_t o_mu  = (size_t)BB * KK + M_GATE * KK;      // after logits + logits_step
    const size_t o_sg  = o_mu + M_DEC * 2;
    const size_t o_rho = o_sg + M_DEC * 2;
    k_heads<<<(unsigned)((M_DEC + 255) / 256), 256, 0, stream>>>(
        ws + P_H2, in(56), in(57), in(58), in(59), out, o_mu, o_sg, o_rho);
}
